// DevignBaseline_82403242541169
// MI455X (gfx1250) — matrix operand order repaired and checked by reading, not compiled, not run on hardware
//
#include <hip/hip_runtime.h>
#include <hip/hip_bf16.h>

typedef __attribute__((ext_vector_type(16))) __bf16 v16bf;
typedef __attribute__((ext_vector_type(8)))  __bf16 bfx8;
typedef __attribute__((ext_vector_type(4)))  __bf16 bfx4;
typedef __attribute__((ext_vector_type(8)))  float  v8f;

#define HID 256
#define NGRAPH 128
#define EMB 128

// ---------------- helpers ----------------

__device__ __forceinline__ __bf16 f2bf(float f) {
    union { float f; unsigned u; } in; in.f = f;
    unsigned u = in.u + 0x7FFFu + ((in.u >> 16) & 1u);   // round-to-nearest-even
    union { unsigned short s; __bf16 b; } out;
    out.s = (unsigned short)(u >> 16);
    return out.b;
}

__device__ __forceinline__ float sigmoidf_(float x) {
    return 1.0f / (1.0f + __expf(-x));
}

// A fragment (16x32 bf16): lane = row (lane&15); halves 0..7 -> K = kk + (lane>>4)*8 + i,
// halves 8..15 -> K = kk+16 + (lane>>4)*8 + i. Two 16-byte loads per lane, no VALU.
__device__ __forceinline__ v16bf load_a_frag(const __bf16* __restrict__ A, int lda,
                                             int rowbase, int kk, int lane) {
    const __bf16* ap = A + (size_t)(rowbase + (lane & 15)) * lda + kk + ((lane >> 4) << 3);
    bfx8 lo = *(const bfx8*)ap;
    bfx8 hi = *(const bfx8*)(ap + 16);
    return __builtin_shufflevector(lo, hi, 0,1,2,3,4,5,6,7,8,9,10,11,12,13,14,15);
}

// B fragment (32x16 bf16), weight stored [N,K] row-major: lane -> col N = ncol0+(lane&15);
// halves 0..7 -> K = kk + (lane>>4)*8 + i, halves 8..15 -> K = kk+16 + (lane>>4)*8 + i. Two 16-byte loads.
__device__ __forceinline__ v16bf load_b_frag(const __bf16* __restrict__ W, int ldb,
                                             int ncol0, int kk, int lane) {
    const __bf16* bp = W + (size_t)(ncol0 + (lane & 15)) * ldb + kk + ((lane >> 4) << 3);
    bfx8 lo = *(const bfx8*)bp;
    bfx8 hi = *(const bfx8*)(bp + 16);
    return __builtin_shufflevector(lo, hi, 0,1,2,3,4,5,6,7,8,9,10,11,12,13,14,15);
}

// ---------------- one-time conversion kernels ----------------

__global__ __launch_bounds__(256) void f32_to_bf16_k(const float* __restrict__ src,
                                                     __bf16* __restrict__ dst, int n4) {
    int i = blockIdx.x * blockDim.x + threadIdx.x;
    if (i >= n4) return;
    float4 v = *(const float4*)(src + (size_t)i * 4);
    bfx4 o; o[0] = f2bf(v.x); o[1] = f2bf(v.y); o[2] = f2bf(v.z); o[3] = f2bf(v.w);
    *(bfx4*)(dst + (size_t)i * 4) = o;
}

// ggnn [3][K][N] -> bf16 transposed [3][N][K] so the GEMM can use the contiguous-K path
__global__ __launch_bounds__(256) void transpose_bf_k(const float* __restrict__ G,
                                                      __bf16* __restrict__ Gt) {
    int idx = blockIdx.x * blockDim.x + threadIdx.x;      // l*65536 + n*256 + k
    if (idx >= 3 * HID * HID) return;
    int k = idx & 255, n = (idx >> 8) & 255, l = idx >> 16;
    Gt[idx] = f2bf(G[((size_t)l << 16) + (k << 8) + n]);
}

// x_bf[n][:] = bf16(embed[x_lex[n]][:])
__global__ __launch_bounds__(256) void gather_embed_k(const float* __restrict__ embed,
                                                      const int* __restrict__ x_lex,
                                                      __bf16* __restrict__ xbf, int N) {
    int i = blockIdx.x * blockDim.x + threadIdx.x;        // N*32 threads, 4 cols each
    int node = i >> 5, q = (i & 31) << 2;
    if (node >= N) return;
    float4 v = *(const float4*)(embed + (size_t)x_lex[node] * EMB + q);
    bfx4 o; o[0] = f2bf(v.x); o[1] = f2bf(v.y); o[2] = f2bf(v.z); o[3] = f2bf(v.w);
    *(bfx4*)(xbf + (size_t)node * EMB + q) = o;
}

// ---------------- generic WMMA GEMM: C = A_bf16 x W_bf16[N,K]^T ----------------
// Each wave owns 4 adjacent N-tiles (one 16x64 C strip): one A-fragment feeds 4
// independent wmma chains -> 2.5 b128 loads per wmma and good load/wmma overlap.
// Requires (N/16) % 4 == 0 (true for N = 256 and 64 here).
template <bool RELU>
__global__ __launch_bounds__(256) void wmma_gemm_k(
    const __bf16* __restrict__ A, int lda,
    const __bf16* __restrict__ W, int ldb,
    const float* __restrict__ bias,
    float* __restrict__ C, __bf16* __restrict__ Cbf, int ldc,
    int M, int N, int K) {
    int lane   = threadIdx.x & 31;
    int wave   = blockIdx.x * (blockDim.x >> 5) + (threadIdx.x >> 5);
    int stride = gridDim.x * (blockDim.x >> 5);
    int quadsN = N >> 6;                       // groups of 4 N-tiles
    int total  = (M >> 4) * quadsN;
    for (int t = wave; t < total; t += stride) {
        int mt  = (t / quadsN) << 4;
        int nt0 = (t % quadsN) << 6;
        v8f acc[4];
#pragma unroll
        for (int u = 0; u < 4; ++u) acc[u] = 0.0f;
        for (int kk = 0; kk < K; kk += 32) {
            v16bf a = load_a_frag(A, lda, mt, kk, lane);
#pragma unroll
            for (int u = 0; u < 4; ++u) {
                v16bf b = load_b_frag(W, ldb, nt0 + (u << 4), kk, lane);
                acc[u] = __builtin_amdgcn_wmma_f32_16x16x32_bf16(
                    false, a, false, b, (short)0, acc[u], false, false);
            }
        }
#pragma unroll
        for (int u = 0; u < 4; ++u) {
            int   col = nt0 + (u << 4) + (lane & 15);
            float bv  = bias ? bias[col] : 0.0f;
#pragma unroll
            for (int r = 0; r < 8; ++r) {
                int    row = mt + r + ((lane >> 4) << 3);
                float  v   = acc[u][r] + bv;
                if (RELU) v = fmaxf(v, 0.0f);
                size_t idx = (size_t)row * ldc + col;
                C[idx] = v;
                if (Cbf) Cbf[idx] = f2bf(v);
            }
        }
    }
}

// ---------------- edge scatter-add (one wave per edge) ----------------
__global__ __launch_bounds__(256) void scatter_add_k(
    const float* __restrict__ m, const int* __restrict__ src,
    const int* __restrict__ dst, float* __restrict__ agg, int E) {
    int gw   = (blockIdx.x * blockDim.x + threadIdx.x) >> 5;
    int lane = threadIdx.x & 31;
    if (gw >= E) return;
    int s = src[gw], d = dst[gw];
    const float* mp = m   + (size_t)s * HID;
    float*       ap = agg + (size_t)d * HID;
#pragma unroll
    for (int j = 0; j < HID / 32; ++j) {
        int c = lane + j * 32;
        __hip_atomic_fetch_add(ap + c, mp[c], __ATOMIC_RELAXED, __HIP_MEMORY_SCOPE_AGENT);
    }
}

// ---------------- fused GRU cell ----------------
// Block = 512 threads = 16 waves, one 16-column group per wave: 6 WMMA accumulators
// per wave (r/z/n x gi/gh), K=256 -> 48 v_wmma per wave. Low register pressure lets
// the scheduler keep all 6 B-fragment loads of a K-step in flight. One barrier
// separates all h_bf reads (GEMM A-frags of every wave) from the in-place h update.
__global__ __launch_bounds__(512) void gru_fused_k(
    const __bf16* __restrict__ aggbf, const __bf16* __restrict__ hbf_in,
    float* __restrict__ h, __bf16* __restrict__ hbf,
    const __bf16* __restrict__ W_ih, const __bf16* __restrict__ W_hh,
    const float* __restrict__ b_ih, const float* __restrict__ b_hh) {
    int lane = threadIdx.x & 31;
    int wid  = threadIdx.x >> 5;           // 0..15 == column group
    int mt   = blockIdx.x << 4;            // 16-row strip
    int nt   = wid << 4;                   // column base (0..240)

    v8f acc[6];
#pragma unroll
    for (int i = 0; i < 6; ++i) acc[i] = 0.0f;

    for (int kk = 0; kk < HID; kk += 32) {
        v16bf aA = load_a_frag(aggbf,  HID, mt, kk, lane);
        v16bf aH = load_a_frag(hbf_in, HID, mt, kk, lane);
#pragma unroll
        for (int q = 0; q < 3; ++q) {      // r, z, n gates
            v16bf bI = load_b_frag(W_ih, HID, nt + q * HID, kk, lane);
            acc[q] = __builtin_amdgcn_wmma_f32_16x16x32_bf16(
                false, aA, false, bI, (short)0, acc[q], false, false);
            v16bf bH = load_b_frag(W_hh, HID, nt + q * HID, kk, lane);
            acc[3 + q] = __builtin_amdgcn_wmma_f32_16x16x32_bf16(
                false, aH, false, bH, (short)0, acc[3 + q], false, false);
        }
    }

    __syncthreads();   // all waves done reading h rows of this strip

    int   col = nt + (lane & 15);
    float bir = b_ih[col], biz = b_ih[col + HID], bin = b_ih[col + 2 * HID];
    float bhr = b_hh[col], bhz = b_hh[col + HID], bhn = b_hh[col + 2 * HID];
#pragma unroll
    for (int r = 0; r < 8; ++r) {
        int    row = mt + r + ((lane >> 4) << 3);
        float  rr  = sigmoidf_(acc[0][r] + bir + acc[3][r] + bhr);
        float  zz  = sigmoidf_(acc[1][r] + biz + acc[4][r] + bhz);
        float  nn  = tanhf(acc[2][r] + bin + rr * (acc[5][r] + bhn));
        size_t idx = (size_t)row * HID + col;
        float  hn  = (1.0f - zz) * nn + zz * h[idx];
        h[idx]   = hn;
        hbf[idx] = f2bf(hn);
    }
}

// ---------------- segment max pool (batch is sorted) ----------------
__global__ __launch_bounds__(256) void pool_max_k(
    const float* __restrict__ h, const int* __restrict__ batch, int N,
    float* __restrict__ pool, __bf16* __restrict__ poolbf) {
    int g = blockIdx.x;      // graph
    int c = threadIdx.x;     // column 0..255
    int lo = 0, hi = N;
    while (lo < hi) { int mid = (lo + hi) >> 1; if (batch[mid] < g) lo = mid + 1; else hi = mid; }
    int s = lo;
    lo = s; hi = N;
    while (lo < hi) { int mid = (lo + hi) >> 1; if (batch[mid] < g + 1) lo = mid + 1; else hi = mid; }
    int e = lo;
    float v = -__builtin_inff();
    for (int n = s; n < e; ++n) v = fmaxf(v, h[(size_t)n * HID + c]);
    pool[(size_t)g * HID + c]   = v;
    poolbf[(size_t)g * HID + c] = f2bf(v);
}

// ---------------- final logits: relu-hidden [G,64] x W2[1,64] ----------------
__global__ void logits_k(const float* __restrict__ t, const float* __restrict__ W2,
                         const float* __restrict__ b2, float* __restrict__ out) {
    int g = blockIdx.x * blockDim.x + threadIdx.x;
    if (g >= NGRAPH) return;
    float s = 0.0f;
    for (int j = 0; j < 64; ++j) s += t[g * 64 + j] * W2[j];
    out[g] = s + b2[0];
}

// ---------------- launch ----------------
extern "C" void kernel_launch(void* const* d_in, const int* in_sizes, int n_in,
                              void* d_out, int out_size, void* d_ws, size_t ws_size,
                              hipStream_t stream) {
    const int*   x_lex  = (const int*)d_in[0];
    const int*   edge   = (const int*)d_in[1];
    const int*   batch  = (const int*)d_in[2];
    const float* embed  = (const float*)d_in[3];   // [VOCAB,128]
    const float* W_proj = (const float*)d_in[4];   // [256,128]
    const float* ggnn   = (const float*)d_in[5];   // [3,256,256]
    const float* W_ih   = (const float*)d_in[6];   // [768,256]
    const float* b_ih   = (const float*)d_in[7];
    const float* W_hh   = (const float*)d_in[8];   // [768,256]
    const float* b_hh   = (const float*)d_in[9];
    const float* W1     = (const float*)d_in[10];  // [64,256]
    const float* b1     = (const float*)d_in[11];
    const float* W2     = (const float*)d_in[12];  // [1,64]
    const float* b2     = (const float*)d_in[13];

    int N = in_sizes[0];          // 50000 (multiple of 16)
    int E = in_sizes[1] / 2;      // 800000
    const int* src = edge;
    const int* dst = edge + E;

    // ---- workspace carve-up (all offsets 16B-aligned) ----
    size_t fN = (size_t)N * HID;
    float*  h      = (float*)d_ws;            // fN
    float*  m      = h + fN;                  // fN
    float*  agg    = m + fN;                  // fN
    float*  t      = agg + fN;                // 128*64
    __bf16* hbf    = (__bf16*)(t + NGRAPH * 64);
    __bf16* aggbf  = hbf + fN;
    __bf16* xbf    = aggbf + fN;              // N*128
    __bf16* wprojb = xbf + (size_t)N * EMB;   // 256*128
    __bf16* gtb    = wprojb + HID * EMB;      // 3*256*256 (transposed)
    __bf16* wihb   = gtb + 3 * HID * HID;     // 768*256
    __bf16* whhb   = wihb + 3 * HID * HID;    // 768*256
    __bf16* w1b    = whhb + 3 * HID * HID;    // 64*256
    __bf16* poolbf = w1b + 64 * HID;          // 128*256

    float* logits = (float*)d_out;            // [128]
    float* pool   = (float*)d_out + NGRAPH;   // [128,256]

    // ---- one-time bf16 weight prep ----
    auto conv = [&](const float* s, __bf16* d, int n) {
        f32_to_bf16_k<<<(n / 4 + 255) / 256, 256, 0, stream>>>(s, d, n / 4);
    };
    conv(W_proj, wprojb, HID * EMB);
    conv(W_ih,   wihb,   3 * HID * HID);
    conv(W_hh,   whhb,   3 * HID * HID);
    conv(W1,     w1b,    64 * HID);
    transpose_bf_k<<<(3 * HID * HID + 255) / 256, 256, 0, stream>>>(ggnn, gtb);
    gather_embed_k<<<(N * 32 + 255) / 256, 256, 0, stream>>>(embed, x_lex, xbf, N);

    int quads      = (N / 16) * (HID / 64);   // 4 N-tiles per wave
    int gemmBlocks = (quads + 7) / 8;         // 8 waves per 256-thread block

    // x = embed[x_lex] @ W_proj.T  -> h (f32) + hbf (bf16)
    wmma_gemm_k<false><<<gemmBlocks, 256, 0, stream>>>(
        xbf, EMB, wprojb, EMB, nullptr, h, hbf, HID, N, HID, EMB);

    for (int L = 0; L < 3; ++L) {
        // m = h @ ggnn[L]  (Gt is [N,K] bf16)
        wmma_gemm_k<false><<<gemmBlocks, 256, 0, stream>>>(
            hbf, HID, gtb + (size_t)L * HID * HID, HID, nullptr,
            m, (__bf16*)nullptr, HID, N, HID, HID);
        // agg = segment_sum(m[src], dst)  (f32 L2 atomics)
        hipMemsetAsync(agg, 0, fN * sizeof(float), stream);
        scatter_add_k<<<(E + 7) / 8, 256, 0, stream>>>(m, src, dst, agg, E);
        // agg -> bf16 for the GRU GEMMs
        f32_to_bf16_k<<<((int)(fN / 4) + 255) / 256, 256, 0, stream>>>(agg, aggbf, (int)(fN / 4));
        // h = GRU(agg, h)  (updates h and hbf in place)
        gru_fused_k<<<N / 16, 512, 0, stream>>>(aggbf, hbf, h, hbf, wihb, whhb, b_ih, b_hh);
    }

    // h_pool = segment_max(h, batch)
    pool_max_k<<<NGRAPH, 256, 0, stream>>>(h, batch, N, pool, poolbf);

    // t = relu(h_pool @ W1.T + b1)   (M=128, N=64, K=256)
    int poolQuads = (NGRAPH / 16) * (64 / 64);
    wmma_gemm_k<true><<<(poolQuads + 7) / 8, 256, 0, stream>>>(
        poolbf, HID, w1b, HID, b1, t, (__bf16*)nullptr, 64, NGRAPH, 64, HID);

    // logits = t @ W2.T + b2
    logits_k<<<1, NGRAPH, 0, stream>>>(t, W2, b2, logits);
}
